// JointSpatioTemporalEncoding_52561809769174
// MI455X (gfx1250) — compile-verified, matrix operand order checked by reading, not run on hardware
//
#include <hip/hip_runtime.h>
#include <hip/hip_bf16.h>

typedef __attribute__((ext_vector_type(16))) _Float16 v16h;
typedef __attribute__((ext_vector_type(8)))  _Float16 v8h;
typedef __attribute__((ext_vector_type(8)))  float    v8f;
typedef __attribute__((ext_vector_type(4)))  int      v4i;

#define B_  2
#define S_  64
#define N_  48
#define D_  128
#define H_  8
#define DH_ 16
#define L_  (S_*N_)     // 3072
#define T_  (B_*S_*N_)  // 6144

// Async global->LDS staging (gfx1250 GLOBAL_LOAD_ASYNC_TO_LDS_B128, ASYNCcnt),
// guarded so the file compiles on toolchains without the builtins.
#if defined(__gfx1250__) && __has_builtin(__builtin_amdgcn_global_load_async_to_lds_b128) && __has_builtin(__builtin_amdgcn_s_wait_asynccnt)
#define USE_ASYNC_LDS 1
#else
#define USE_ASYNC_LDS 0
#endif

#define GLOBAL_AS __attribute__((address_space(1)))
#define LDS_AS    __attribute__((address_space(3)))

// ---------------------------------------------------------------------------
// WMMA helpers (CDNA5 gfx1250, wave32). A: 16x32 f16, B: 32x16 f16, C/D: 16x16 f32.
// A layout (ISA 7.12.2): lane l holds row M=l&15; halves v0..7 -> K = kb+0..7,
// halves v8..15 -> K = kb+16..23, where kb = (l<16) ? 0 : 8.
// B assumed to mirror A with N = l&15 over the same K pattern; weights are
// pre-packed into exactly that layout so each lane reads 16 contiguous halves.
// C layout: vgpr r, lane l -> M = r + 8*(l>>4), N = l&15.
// ---------------------------------------------------------------------------
__device__ __forceinline__ v8f wmma_f16f32(v16h a, v16h b, v8f c) {
  return __builtin_amdgcn_wmma_f32_16x16x32_f16(false, a, false, b, (short)0, c,
                                                false, false);
}

__device__ __forceinline__ v16h make_a(v8h lo, v8h hi) {
  v16h a;
#pragma unroll
  for (int i = 0; i < 8; ++i) { a[i] = lo[i]; a[i + 8] = hi[i]; }
  return a;
}

__device__ __forceinline__ v16h make_a_padhi(v8h lo) {
  v16h a;
#pragma unroll
  for (int i = 0; i < 8; ++i) { a[i] = lo[i]; a[i + 8] = (_Float16)0.0f; }
  return a;
}

// ---------------------------------------------------------------------------
// Weight packing: W(Dout x Din) with element stride es and offset eo
// (handles conv taps: element (dout,din) = src[(dout*Din+din)*es + eo]).
// Packed as tiles (kc over Din/32, ct over Dout/16); tile = 512 halves;
// lane l stores 16 halves at tile*512 + l*16 matching the B-operand pattern.
// ---------------------------------------------------------------------------
__global__ void pack_w_kernel(const float* __restrict__ src, _Float16* __restrict__ dst,
                              int Dout, int Din, int es, int eo) {
  int id = blockIdx.x * blockDim.x + threadIdx.x;
  if (id >= Dout * Din) return;
  int tile   = id >> 9;
  int within = id & 511;
  int l = within >> 4, v = within & 15;
  int colTiles = Dout >> 4;
  int kc = tile / colTiles, ct = tile - kc * colTiles;
  int dout = ct * 16 + (l & 15);
  int base = (l < 16) ? 0 : 8;
  int kk   = (v < 8) ? (base + v) : (base + 16 + (v - 8));
  int din  = kc * 32 + kk;
  dst[id] = (_Float16)src[(size_t)(dout * Din + din) * es + eo];
}

// ---------------------------------------------------------------------------
// Generic WMMA GEMM, Nout = 128 fixed: out = epi(A(M x K) @ Wpacked + bias).
// One 16-row x 64-col strip per wave (4 accumulators): the A tile is loaded
// once per K-chunk and reused by 4 WMMAs; K is a template param so the whole
// K-loop unrolls into immediate-offset global_load_b128s.
// epi: 0 = none, 1 = relu, 2 = sigmoid.
// ---------------------------------------------------------------------------
template <int K>
__global__ void gemm_epi_kernel(const _Float16* __restrict__ A,
                                const _Float16* __restrict__ Bp,
                                const float* __restrict__ bias,
                                float* __restrict__ outF, _Float16* __restrict__ outH,
                                int M, int epi, int accumulate) {
  constexpr int colTiles = D_ / 16;  // 8
  int lane = threadIdx.x & 31;
  int wave = threadIdx.x >> 5;
  int tile  = blockIdx.x * (blockDim.x >> 5) + wave;
  int total = (M >> 4) * 2;          // 2 column groups of 4 tiles
  if (tile >= total) return;
  int rt = tile >> 1, cg = tile & 1, ct0 = cg * 4;
  int kb  = (lane >> 4) << 3;
  int cid = lane & 15;
  const _Float16* ar = A + (size_t)(rt * 16 + cid) * K + kb;
  const _Float16* wm = Bp + (size_t)ct0 * 512 + (lane << 4);
  v8f acc[4];
#pragma unroll
  for (int j = 0; j < 4; ++j)
#pragma unroll
    for (int r = 0; r < 8; ++r) acc[j][r] = 0.f;
#pragma unroll
  for (int kc = 0; kc < K / 32; ++kc) {
    v8h lo = *(const v8h*)(ar + kc * 32);
    v8h hi = *(const v8h*)(ar + kc * 32 + 16);
    v16h a = make_a(lo, hi);
    const _Float16* wk = wm + (size_t)kc * colTiles * 512;
#pragma unroll
    for (int j = 0; j < 4; ++j) {
      v16h b = *(const v16h*)(wk + j * 512);
      acc[j] = wmma_f16f32(a, b, acc[j]);
    }
  }
  int rowBase = rt * 16 + ((lane >> 4) << 3);
#pragma unroll
  for (int j = 0; j < 4; ++j) {
    int col  = (ct0 + j) * 16 + cid;
    float bv = bias ? bias[col] : 0.f;
#pragma unroll
    for (int r = 0; r < 8; ++r) {
      float v = acc[j][r] + bv;
      if (epi == 1)      v = fmaxf(v, 0.f);
      else if (epi == 2) v = 1.f / (1.f + __expf(-v));
      size_t idx = (size_t)(rowBase + r) * D_ + col;
      if (outF) { if (accumulate) outF[idx] += v; else outF[idx] = v; }
      if (outH) outH[idx] = (_Float16)v;
    }
  }
}

// ---------------------------------------------------------------------------
// Temporal conv branch. One 16-token x 64-channel strip per wave; the shifted
// A tile is loaded once per (tap, K-chunk) and feeds 4 WMMAs. Tokens in a
// tile share (b,s) since N=48 is a multiple of 16, so tap validity
// (0 <= s+d < S) is wave-uniform. Shift by d in s == shift rows by d*N_.
// outT = (sum_k relu(conv_k))/4 + xp.
// ---------------------------------------------------------------------------
__global__ void conv_branch_kernel(const _Float16* __restrict__ xph,
                                   const _Float16* __restrict__ Wp,
                                   const float* __restrict__ b1, const float* __restrict__ b3,
                                   const float* __restrict__ b5, const float* __restrict__ b7,
                                   const float* __restrict__ xpf, float* __restrict__ outT) {
  constexpr int colTiles = D_ / 16;  // 8
  int lane = threadIdx.x & 31;
  int wave = threadIdx.x >> 5;
  int tile = blockIdx.x * (blockDim.x >> 5) + wave;
  const int totalTiles = (T_ / 16) * 2;  // 768
  if (tile >= totalTiles) return;
  int rt = tile >> 1, cg = tile & 1, ct0 = cg * 4;
  int m0 = rt * 16;
  int s  = (m0 / N_) % S_;
  int kb  = (lane >> 4) << 3;
  int cid = lane & 15;
  const float* biases[4] = {b1, b3, b5, b7};
  const _Float16* aBase = xph + (size_t)(m0 + cid) * D_ + kb;
  const _Float16* wBase = Wp + (size_t)ct0 * 512 + (lane << 4);
  v8f accF[4];
#pragma unroll
  for (int j = 0; j < 4; ++j)
#pragma unroll
    for (int r = 0; r < 8; ++r) accF[j][r] = 0.f;
  int matIdx = 0;
#pragma unroll
  for (int ci = 0; ci < 4; ++ci) {
    const int ksz = 2 * ci + 1;
    v8f accC[4];
#pragma unroll
    for (int j = 0; j < 4; ++j)
#pragma unroll
      for (int r = 0; r < 8; ++r) accC[j][r] = 0.f;
#pragma unroll
    for (int t = 0; t < ksz; ++t) {
      int d  = t - ci;          // tap shift (ksz/2 == ci)
      int ss = s + d;
      if (ss >= 0 && ss < S_) {
        const _Float16* ar = aBase + (ptrdiff_t)(d * N_) * D_;
        const _Float16* wm = wBase + (size_t)(matIdx + t) * D_ * D_;
#pragma unroll
        for (int kc = 0; kc < 4; ++kc) {
          v16h a = make_a(*(const v8h*)(ar + kc * 32),
                          *(const v8h*)(ar + kc * 32 + 16));
          const _Float16* wk = wm + (size_t)kc * colTiles * 512;
#pragma unroll
          for (int j = 0; j < 4; ++j) {
            v16h b = *(const v16h*)(wk + j * 512);
            accC[j] = wmma_f16f32(a, b, accC[j]);
          }
        }
      }
    }
    matIdx += ksz;
#pragma unroll
    for (int j = 0; j < 4; ++j) {
      float bv = biases[ci][(ct0 + j) * 16 + cid];
#pragma unroll
      for (int r = 0; r < 8; ++r) accF[j][r] += fmaxf(accC[j][r] + bv, 0.f);
    }
  }
  int rowBase = m0 + ((lane >> 4) << 3);
#pragma unroll
  for (int j = 0; j < 4; ++j) {
    int col = (ct0 + j) * 16 + cid;
#pragma unroll
    for (int r = 0; r < 8; ++r) {
      size_t idx = (size_t)(rowBase + r) * D_ + col;
      outT[idx] = accF[j][r] * 0.25f + xpf[idx];
    }
  }
}

// ---------------------------------------------------------------------------
// Flash attention: one wave per (b, h, 16-row tile). Stream columns in chunks
// of 32; scores via two WMMAs (DH=16 zero-padded to K=32); P relayout through
// wave-private LDS (same-wave DS ops are in-order); PV via one WMMA (K=32).
// V staging uses GLOBAL_LOAD_ASYNC_TO_LDS_B128 when available, issued at the
// top of the iteration and waited (s_wait_asynccnt 0) only right before the
// PV operand read, overlapping with the score WMMAs + softmax VALU.
// ---------------------------------------------------------------------------
__global__ void attn_kernel(const _Float16* __restrict__ qh, const _Float16* __restrict__ kh,
                            const _Float16* __restrict__ vh, _Float16* __restrict__ aoH) {
  __shared__ _Float16 ldsP[8][16 * 32];
  __shared__ _Float16 ldsV[8][32 * 16];
  int lane = threadIdx.x & 31;
  int w    = threadIdx.x >> 5;
  const int rowTiles = L_ / 16;  // 192
  int wid = blockIdx.x * (blockDim.x >> 5) + w;
  if (wid >= B_ * H_ * rowTiles) return;
  int rt = wid % rowTiles;
  int bh = wid / rowTiles;
  int h = bh % H_, b = bh / H_;
  size_t bL = (size_t)b * L_;
  const float SC = 0.25f;  // 1/sqrt(DH)
  int kb  = (lane >> 4) << 3;
  int cid = lane & 15;

  // Hoisted per-lane base pointers; per iteration only += c0*D_.
  const _Float16* qp   = qh + (bL + rt * 16 + cid) * D_ + h * DH_ + kb;
  const _Float16* kRow = kh + (bL + cid) * D_ + h * DH_ + kb;
  const _Float16* vRow = vh + (bL + lane) * D_ + h * DH_;
  _Float16* ldsVLane = &ldsV[w][lane * 16];

  v16h aQ = make_a_padhi(*(const v8h*)qp);

  float Mr[8], Sr[8];
  v8f O;
#pragma unroll
  for (int r = 0; r < 8; ++r) { Mr[r] = -1e30f; Sr[r] = 0.f; O[r] = 0.f; }

  for (int c0 = 0; c0 < L_; c0 += 32) {
    size_t coff = (size_t)c0 * D_;
    // stage V[c0..c0+31][h][0..15] into LDS, one token row per lane
#if USE_ASYNC_LDS
    __builtin_amdgcn_global_load_async_to_lds_b128(
        (GLOBAL_AS v4i*)(vRow + coff), (LDS_AS v4i*)ldsVLane, 0, 0);
    __builtin_amdgcn_global_load_async_to_lds_b128(
        (GLOBAL_AS v4i*)(vRow + coff + 8), (LDS_AS v4i*)(ldsVLane + 8), 0, 0);
#else
    {
      v8h va = *(const v8h*)(vRow + coff);
      v8h vb = *(const v8h*)(vRow + coff + 8);
      *(v8h*)ldsVLane       = va;
      *(v8h*)(ldsVLane + 8) = vb;
    }
#endif
    // scores: S = Q @ K^T for 2 x 16 columns
    v16h bK0 = make_a_padhi(*(const v8h*)(kRow + coff));
    v16h bK1 = make_a_padhi(*(const v8h*)(kRow + coff + (size_t)16 * D_));
    v8f z;
#pragma unroll
    for (int r = 0; r < 8; ++r) z[r] = 0.f;
    v8f s0v = wmma_f16f32(aQ, bK0, z);
    v8f s1v = wmma_f16f32(aQ, bK1, z);

#pragma unroll
    for (int r = 0; r < 8; ++r) {
      float s0 = s0v[r] * SC, s1 = s1v[r] * SC;
      float mx = fmaxf(s0, s1);
      for (int o = 8; o > 0; o >>= 1) mx = fmaxf(mx, __shfl_xor(mx, o, 16));
      float mnew  = fmaxf(Mr[r], mx);
      float alpha = __expf(Mr[r] - mnew);
      float p0 = __expf(s0 - mnew), p1 = __expf(s1 - mnew);
      float rs = p0 + p1;
      for (int o = 8; o > 0; o >>= 1) rs += __shfl_xor(rs, o, 16);
      Sr[r] = Sr[r] * alpha + rs;
      Mr[r] = mnew;
      O[r] *= alpha;
      int prow = r + ((lane >> 4) << 3);
      ldsP[w][prow * 32 + cid]      = (_Float16)p0;
      ldsP[w][prow * 32 + cid + 16] = (_Float16)p1;
    }
#if USE_ASYNC_LDS
    __builtin_amdgcn_s_wait_asynccnt(0);
#endif
    asm volatile("" ::: "memory");  // same-wave DS ops are in-order; stop compiler reordering

    // P as A-operand (16x32 f16)
    v16h aP;
    {
      int row = lane & 15;
      v8h lo = *(const v8h*)&ldsP[w][row * 32 + kb];
      v8h hi = *(const v8h*)&ldsP[w][row * 32 + kb + 16];
      aP = make_a(lo, hi);
    }
    // V as B-operand (32x16 f16): B[k=col][n=dh]
    v16h bV;
#pragma unroll
    for (int i = 0; i < 8; ++i) {
      bV[i]     = ldsV[w][(kb + i) * 16 + cid];
      bV[i + 8] = ldsV[w][(kb + 16 + i) * 16 + cid];
    }
    O = wmma_f16f32(aP, bV, O);
    asm volatile("" ::: "memory");
  }
#pragma unroll
  for (int r = 0; r < 8; ++r) {
    float oo = O[r] / Sr[r];
    int row  = rt * 16 + r + ((lane >> 4) << 3);
    aoH[(bL + row) * D_ + h * DH_ + cid] = (_Float16)oo;
  }
}

// ---------------------------------------------------------------------------
// Small VALU kernels
// ---------------------------------------------------------------------------
__global__ void prep_kernel(const float* __restrict__ x, const float* __restrict__ sp,
                            const float* __restrict__ tp, float* __restrict__ xpf,
                            _Float16* __restrict__ xph) {
  int id = blockIdx.x * blockDim.x + threadIdx.x;
  if (id >= T_ * D_) return;
  int d = id & (D_ - 1);
  int tok = id >> 7;
  int n = tok % N_;
  int s = (tok / N_) % S_;
  float v = x[id] + sp[n * D_ + d] + tp[s * D_ + d];
  xpf[id] = v;
  xph[id] = (_Float16)v;
}

__global__ void adj_kernel(const float* __restrict__ adj, float* __restrict__ nadjP) {
  __shared__ float a1[N_ * N_], a2[N_ * N_];
  int t = threadIdx.x;
  if (t < N_) {
    float dsum = 0.f;
    for (int n = 0; n < N_; ++n) dsum += adj[t * N_ + n];
    if (dsum == 0.f) dsum = 1.f;
    float inv = 1.f / dsum;
    for (int n = 0; n < N_; ++n) {
      float v = adj[t * N_ + n] * inv;
      a1[t * N_ + n] = v;
      nadjP[t * N_ + n] = v;
    }
  }
  __syncthreads();
  for (int id = t; id < N_ * N_; id += blockDim.x) {
    int m = id / N_, n = id % N_;
    float s2 = 0.f;
    for (int k = 0; k < N_; ++k) s2 += a1[m * N_ + k] * a1[k * N_ + n];
    a2[id] = s2;
    nadjP[N_ * N_ + id] = s2;
  }
  __syncthreads();
  for (int id = t; id < N_ * N_; id += blockDim.x) {
    int m = id / N_, n = id % N_;
    float s3 = 0.f;
    for (int k = 0; k < N_; ++k) s3 += a2[m * N_ + k] * a1[k * N_ + n];
    nadjP[2 * N_ * N_ + id] = s3;
  }
}

__global__ void nodemix_kernel(const float* __restrict__ xpf, const float* __restrict__ nadjP,
                               _Float16* __restrict__ y0, _Float16* __restrict__ y1,
                               _Float16* __restrict__ y2) {
  int id = blockIdx.x * blockDim.x + threadIdx.x;
  if (id >= T_ * D_) return;
  int d   = id & (D_ - 1);
  int tok = id >> 7;
  int m   = tok % N_;
  size_t base = (size_t)(tok - m) * D_ + d;
  const float* A0 = nadjP;
  const float* A1 = nadjP + N_ * N_;
  const float* A2 = nadjP + 2 * N_ * N_;
  float s0 = 0.f, s1 = 0.f, s2 = 0.f;
  for (int k = 0; k < N_; ++k) {
    float xv = xpf[base + (size_t)k * D_];
    s0 += A0[m * N_ + k] * xv;
    s1 += A1[m * N_ + k] * xv;
    s2 += A2[m * N_ + k] * xv;
  }
  y0[id] = (_Float16)s0;
  y1[id] = (_Float16)s1;
  y2[id] = (_Float16)s2;
}

__global__ void ln_kernel(const float* __restrict__ in, const float* __restrict__ g,
                          const float* __restrict__ bta, float* __restrict__ outF,
                          _Float16* __restrict__ outH) {
  int lane = threadIdx.x & 31;
  int wave = threadIdx.x >> 5;
  int row = blockIdx.x * (blockDim.x >> 5) + wave;
  if (row >= T_) return;
  const float* p = in + (size_t)row * D_;
  float v[4];
#pragma unroll
  for (int i = 0; i < 4; ++i) v[i] = p[i * 32 + lane];
  float s = v[0] + v[1] + v[2] + v[3];
  for (int o = 16; o > 0; o >>= 1) s += __shfl_xor(s, o, 32);
  float mean = s * (1.f / D_);
  float q = 0.f;
#pragma unroll
  for (int i = 0; i < 4; ++i) { float dv = v[i] - mean; q += dv * dv; }
  for (int o = 16; o > 0; o >>= 1) q += __shfl_xor(q, o, 32);
  float rstd = rsqrtf(q * (1.f / D_) + 1e-5f);
#pragma unroll
  for (int i = 0; i < 4; ++i) {
    int c = i * 32 + lane;
    float ov = (v[i] - mean) * rstd * g[c] + bta[c];
    size_t idx = (size_t)row * D_ + c;
    if (outF) outF[idx] = ov;
    if (outH) outH[idx] = (_Float16)ov;
  }
}

__global__ void inter_kernel(const float* __restrict__ tf, const float* __restrict__ sf,
                             const float* __restrict__ w, float* __restrict__ out) {
  int id = blockIdx.x * blockDim.x + threadIdx.x;
  if (id >= T_ * D_) return;
  int d = id & (D_ - 1);
  int h = d >> 4, i = d & 15;
  int tok = id >> 7;
  const float* srow = sf + (size_t)tok * D_ + h * DH_;
  const float* wrow = w + (h * DH_ + i) * DH_;
  float s = 0.f;
#pragma unroll
  for (int j = 0; j < 16; ++j) s += wrow[j] * srow[j];
  out[id] = tf[id] * s;
}

__global__ void cat_kernel(const float* __restrict__ a, const _Float16* __restrict__ xph,
                           _Float16* __restrict__ cat) {
  int id = blockIdx.x * blockDim.x + threadIdx.x;
  if (id >= T_ * 2 * D_) return;
  int c = id & (2 * D_ - 1);
  int row = id >> 8;
  cat[id] = (c < D_) ? (_Float16)a[(size_t)row * D_ + c]
                     : xph[(size_t)row * D_ + (c - D_)];
}

__global__ void ew_scale_add(const float* __restrict__ a, const float* __restrict__ xp,
                             float sc, float* __restrict__ out) {
  int id = blockIdx.x * blockDim.x + threadIdx.x;
  if (id >= T_ * D_) return;
  out[id] = a[id] * sc + xp[id];
}

__global__ void ew_comb(const float* __restrict__ ig, const float* __restrict__ cg,
                        const float* __restrict__ inter, const float* __restrict__ cross,
                        float* __restrict__ comb) {
  int id = blockIdx.x * blockDim.x + threadIdx.x;
  if (id >= T_ * D_) return;
  comb[id] = ig[id] * inter[id] + cg[id] * cross[id];
}

__global__ void ew_final(const float* __restrict__ fg, const float* __restrict__ comb,
                         const float* __restrict__ xp, float* __restrict__ out) {
  int id = blockIdx.x * blockDim.x + threadIdx.x;
  if (id >= T_ * D_) return;
  float f = fg[id];
  out[id] = f * comb[id] + (1.f - f) * xp[id];
}

// ---------------------------------------------------------------------------
// Host orchestration
// ---------------------------------------------------------------------------
extern "C" void kernel_launch(void* const* d_in, const int* in_sizes, int n_in,
                              void* d_out, int out_size, void* d_ws, size_t ws_size,
                              hipStream_t stream) {
  (void)in_sizes; (void)n_in; (void)out_size; (void)ws_size;
  const float* x       = (const float*)d_in[0];
  const float* adjm    = (const float*)d_in[1];
  const float* sposp   = (const float*)d_in[2];
  const float* tposp   = (const float*)d_in[3];
  const float* inter_w = (const float*)d_in[4];
  const float* conv_w1 = (const float*)d_in[5];
  const float* conv_b1 = (const float*)d_in[6];
  const float* conv_w3 = (const float*)d_in[7];
  const float* conv_b3 = (const float*)d_in[8];
  const float* conv_w5 = (const float*)d_in[9];
  const float* conv_b5 = (const float*)d_in[10];
  const float* conv_w7 = (const float*)d_in[11];
  const float* conv_b7 = (const float*)d_in[12];
  const float* sp_w0 = (const float*)d_in[13];
  const float* sp_b0 = (const float*)d_in[14];
  const float* sp_w1 = (const float*)d_in[15];
  const float* sp_b1 = (const float*)d_in[16];
  const float* sp_w2 = (const float*)d_in[17];
  const float* sp_b2 = (const float*)d_in[18];
  const float* attn_in_w  = (const float*)d_in[19];
  const float* attn_in_b  = (const float*)d_in[20];
  const float* attn_out_w = (const float*)d_in[21];
  const float* attn_out_b = (const float*)d_in[22];
  const float* gsp_w = (const float*)d_in[23];
  const float* gsp_b = (const float*)d_in[24];
  const float* gtm_w = (const float*)d_in[25];
  const float* gtm_b = (const float*)d_in[26];
  const float* gfu_w = (const float*)d_in[27];
  const float* gfu_b = (const float*)d_in[28];
  const float* ln1_g = (const float*)d_in[29];
  const float* ln1_b = (const float*)d_in[30];
  const float* ln2_g = (const float*)d_in[31];
  const float* ln2_b = (const float*)d_in[32];
  const float* ln3_g = (const float*)d_in[33];
  const float* ln3_b = (const float*)d_in[34];

  char* ws = (char*)d_ws;
  size_t off = 0;
  auto alloc = [&](size_t bytes) -> void* {
    void* p = ws + off;
    off = (off + bytes + 255) & ~(size_t)255;
    return p;
  };
  const size_t TD = (size_t)T_ * D_;

  float*    xpF       = (float*)alloc(TD * 4);
  _Float16* xpH       = (_Float16*)alloc(TD * 2);
  float*    taccF     = (float*)alloc(TD * 4);
  float*    temporalF = (float*)alloc(TD * 4);
  _Float16* temporalH = (_Float16*)alloc(TD * 2);
  float*    spatialF  = (float*)alloc(TD * 4);
  _Float16* spatialH  = (_Float16*)alloc(TD * 2);
  _Float16* y0H       = (_Float16*)alloc(TD * 2);
  _Float16* y1H       = (_Float16*)alloc(TD * 2);
  _Float16* y2H       = (_Float16*)alloc(TD * 2);
  float*    saccF     = (float*)alloc(TD * 4);
  _Float16* qH        = (_Float16*)alloc(TD * 2);
  _Float16* kH        = (_Float16*)alloc(TD * 2);
  _Float16* vH        = (_Float16*)alloc(TD * 2);
  _Float16* aoH       = (_Float16*)alloc(TD * 2);
  float*    crossF    = (float*)alloc(TD * 4);
  float*    interF    = (float*)alloc(TD * 4);
  float*    combF     = (float*)alloc(TD * 4);
  float*    igF       = (float*)alloc(TD * 4);   // later reused for fg
  float*    cgF       = (float*)alloc(TD * 4);
  _Float16* catH      = (_Float16*)alloc(TD * 2 * 2);  // T x 256 halves
  float*    nadjP     = (float*)alloc(3 * N_ * N_ * 4);
  _Float16* convP = (_Float16*)alloc((size_t)16 * D_ * D_ * 2);
  _Float16* sp0P  = (_Float16*)alloc((size_t)D_ * D_ * 2);
  _Float16* sp1P  = (_Float16*)alloc((size_t)D_ * D_ * 2);
  _Float16* sp2P  = (_Float16*)alloc((size_t)D_ * D_ * 2);
  _Float16* wqP   = (_Float16*)alloc((size_t)D_ * D_ * 2);
  _Float16* wkP   = (_Float16*)alloc((size_t)D_ * D_ * 2);
  _Float16* wvP   = (_Float16*)alloc((size_t)D_ * D_ * 2);
  _Float16* woP   = (_Float16*)alloc((size_t)D_ * D_ * 2);
  _Float16* gspP  = (_Float16*)alloc((size_t)D_ * 2 * D_ * 2);
  _Float16* gtmP  = (_Float16*)alloc((size_t)D_ * 2 * D_ * 2);
  _Float16* gfuP  = (_Float16*)alloc((size_t)D_ * 2 * D_ * 2);

  const int ewBlocks  = (int)((TD + 255) / 256);
  const int catBlocks = (int)((TD * 2 + 255) / 256);
  const int lnBlocks  = T_ / 8;

  auto pack = [&](const float* src, _Float16* dst, int Dout, int Din, int es, int eo) {
    int n = Dout * Din;
    pack_w_kernel<<<(n + 255) / 256, 256, 0, stream>>>(src, dst, Dout, Din, es, eo);
  };
  auto gemm = [&](const _Float16* A, const _Float16* Bp, const float* bias,
                  float* oF, _Float16* oH, int M, int K, int epi, int accum) {
    int tiles  = (M / 16) * 2;
    int blocks = (tiles + 7) / 8;
    if (K == 128)
      gemm_epi_kernel<128><<<blocks, 256, 0, stream>>>(A, Bp, bias, oF, oH, M, epi, accum);
    else
      gemm_epi_kernel<256><<<blocks, 256, 0, stream>>>(A, Bp, bias, oF, oH, M, epi, accum);
  };

  // 0) positional add + fp16 copy; adjacency powers
  prep_kernel<<<ewBlocks, 256, 0, stream>>>(x, sposp, tposp, xpF, xpH);
  adj_kernel<<<1, 256, 0, stream>>>(adjm, nadjP);

  // 1) pack all weights into the WMMA B-operand layout (fp16)
  {
    const float* cw[4] = {conv_w1, conv_w3, conv_w5, conv_w7};
    int mat = 0;
    for (int ci = 0; ci < 4; ++ci) {
      int ksz = 2 * ci + 1;
      for (int t = 0; t < ksz; ++t)
        pack(cw[ci], convP + (size_t)(mat + t) * D_ * D_, D_, D_, ksz, t);
      mat += ksz;
    }
  }
  pack(sp_w0, sp0P, D_, D_, 1, 0);
  pack(sp_w1, sp1P, D_, D_, 1, 0);
  pack(sp_w2, sp2P, D_, D_, 1, 0);
  pack(attn_in_w,               wqP, D_, D_, 1, 0);
  pack(attn_in_w + D_ * D_,     wkP, D_, D_, 1, 0);
  pack(attn_in_w + 2 * D_ * D_, wvP, D_, D_, 1, 0);
  pack(attn_out_w, woP, D_, D_, 1, 0);
  pack(gsp_w, gspP, D_, 2 * D_, 1, 0);
  pack(gtm_w, gtmP, D_, 2 * D_, 1, 0);
  pack(gfu_w, gfuP, D_, 2 * D_, 1, 0);

  // 2) temporal branch: convs (WMMA) + residual, then LN1
  conv_branch_kernel<<<96, 256, 0, stream>>>(xpH, convP, conv_b1, conv_b3, conv_b5,
                                             conv_b7, xpF, taccF);
  ln_kernel<<<lnBlocks, 256, 0, stream>>>(taccF, ln1_g, ln1_b, temporalF, temporalH);

  // 3) spatial branch: node mixes (VALU) then 3 accumulated relu-GEMMs, LN2
  nodemix_kernel<<<ewBlocks, 256, 0, stream>>>(xpF, nadjP, y0H, y1H, y2H);
  gemm(y0H, sp0P, sp_b0, saccF, nullptr, T_, D_, 1, 0);
  gemm(y1H, sp1P, sp_b1, saccF, nullptr, T_, D_, 1, 1);
  gemm(y2H, sp2P, sp_b2, saccF, nullptr, T_, D_, 1, 1);
  ew_scale_add<<<ewBlocks, 256, 0, stream>>>(saccF, xpF, 1.f / 3.f, taccF);
  ln_kernel<<<lnBlocks, 256, 0, stream>>>(taccF, ln2_g, ln2_b, spatialF, spatialH);

  // 4) per-head bilinear inter gate
  inter_kernel<<<ewBlocks, 256, 0, stream>>>(temporalF, spatialF, inter_w, interF);

  // 5) attention: QKV projections (WMMA), flash attention, output projection
  gemm(temporalH, wqP, attn_in_b,          nullptr, qH, T_, D_, 0, 0);
  gemm(spatialH,  wkP, attn_in_b + D_,     nullptr, kH, T_, D_, 0, 0);
  gemm(spatialH,  wvP, attn_in_b + 2 * D_, nullptr, vH, T_, D_, 0, 0);
  attn_kernel<<<384, 256, 0, stream>>>(qH, kH, vH, aoH);
  gemm(aoH, woP, attn_out_b, crossF, nullptr, T_, D_, 0, 0);

  // 6) gates (K=256 WMMA GEMMs with sigmoid epilogue) + combines
  cat_kernel<<<catBlocks, 256, 0, stream>>>(interF, xpH, catH);
  gemm(catH, gspP, gsp_b, igF, nullptr, T_, 2 * D_, 2, 0);
  cat_kernel<<<catBlocks, 256, 0, stream>>>(crossF, xpH, catH);
  gemm(catH, gtmP, gtm_b, cgF, nullptr, T_, 2 * D_, 2, 0);
  ew_comb<<<ewBlocks, 256, 0, stream>>>(igF, cgF, interF, crossF, combF);
  cat_kernel<<<catBlocks, 256, 0, stream>>>(combF, xpH, catH);
  gemm(catH, gfuP, gfu_b, igF, nullptr, T_, 2 * D_, 2, 0);  // fg in igF
  ew_final<<<ewBlocks, 256, 0, stream>>>(igF, combF, xpF, taccF);

  // 7) final LN -> d_out (fp32)
  ln_kernel<<<lnBlocks, 256, 0, stream>>>(taccF, ln3_g, ln3_b, (float*)d_out, nullptr);
}
